// LSTMDecoder_53257594470436
// MI455X (gfx1250) — matrix operand order repaired and checked by reading, not compiled, not run on hardware
//
#include <hip/hip_runtime.h>
#include <hip/hip_bf16.h>
#include <math.h>

typedef __bf16 bf16_t;
typedef __attribute__((ext_vector_type(16))) __bf16 v16bf;
typedef __attribute__((ext_vector_type(8)))  __bf16 v8bf;
typedef __attribute__((ext_vector_type(8)))  float  v8f;

#define D_NI   512
#define D_NH   1024
#define D_NZ   256
#define D_LNZ  64
#define D_V    32000
#define D_SEQ  128
#define D_SRC  128
#define D_B    16
#define D_G4H  4096   /* 4*NH  */
#define D_TB   2048   /* SEQ*B */
#define D_XK   768    /* NI+NZ */
#define D_2H   2048

// ---------------------------------------------------------------------------
// WMMA helpers (bf16 A/B, f32 accumulate) -- CDNA5 v_wmma_f32_16x16x32_bf16
// ---------------------------------------------------------------------------
__device__ __forceinline__ v8f wmma_bf16f32(v16bf a, v16bf b, v8f c) {
  return __builtin_amdgcn_wmma_f32_16x16x32_bf16(
      /*neg_a=*/false, a, /*neg_b=*/false, b,
      /*c_mod=*/(short)0, c, /*reuse_a=*/false, /*reuse_b=*/false);
}

// A-matrix 16x32 bf16, per ISA layout: lane holds row (lane&15); half-lane hl
// holds K = {hl*8 .. hl*8+7} U {16+hl*8 .. 16+hl*8+7}  -> two b128 loads.
__device__ __forceinline__ v16bf load_a16x32(const bf16_t* rowk, int hl) {
  v8bf lo = *(const v8bf*)(rowk + hl * 8);
  v8bf hi = *(const v8bf*)(rowk + 16 + hl * 8);
  v16bf a;
#pragma unroll
  for (int e = 0; e < 8; ++e) { a[e] = lo[e]; a[8 + e] = hi[e]; }
  return a;
}

// B-matrix 32x16 bf16: lane holds column (lane&15) = weight row; half-lane hl
// holds K = {hl*8 .. hl*8+7} U {16+hl*8 .. 16+hl*8+7}  -> two b128 loads.
__device__ __forceinline__ v16bf load_b32x16(const bf16_t* rowk, int hl) {
  v8bf lo = *(const v8bf*)(rowk + hl * 8);
  v8bf hi = *(const v8bf*)(rowk + 16 + hl * 8);
  v16bf b;
#pragma unroll
  for (int e = 0; e < 8; ++e) { b[e] = lo[e]; b[8 + e] = hi[e]; }
  return b;
}

__device__ __forceinline__ float sigf(float x) { return 1.0f / (1.0f + __expf(-x)); }

// ---------------------------------------------------------------------------
// Generic bf16 WMMA GEMM: out[M,N](f32) = A[M,K] * W[N,K]^T (+ bias1 + bias2)
// block = 128 threads (4 waves along N); wave tile = (MT*16)M x (NT*16)N.
// Manual unroll-by-2 ping-pong: two explicit operand register sets alternate
// load/consume with NO rotation copies; each set's loads are issued a full
// WMMA burst before consumption, so waits are partial and latency is covered.
// Requires: M % (MT*16)==0, N % (NT*64)==0, (K/32) even (true at call sites).
// ---------------------------------------------------------------------------
template <int MT, int NT>
__global__ __launch_bounds__(128) void gemm_bf16_wmma(
    float* __restrict__ out, const bf16_t* __restrict__ A,
    const bf16_t* __restrict__ W, const float* __restrict__ bias1,
    const float* __restrict__ bias2, int M, int N, int K)
{
  const int lane = threadIdx.x & 31;
  const int wave = threadIdx.x >> 5;
  const int row  = lane & 15;
  const int hl   = lane >> 4;
  const int m0   = blockIdx.y * (MT * 16);
  const int n0   = (blockIdx.x * 4 + wave) * (NT * 16);

  const bf16_t* arow[MT];
#pragma unroll
  for (int i = 0; i < MT; ++i) arow[i] = A + (size_t)(m0 + i * 16 + row) * K;
  const bf16_t* wrow[NT];
#pragma unroll
  for (int j = 0; j < NT; ++j) wrow[j] = W + (size_t)(n0 + j * 16 + row) * K;

  v8f acc[MT][NT] = {};
  v16bf a0[MT], b0[NT], a1[MT], b1[NT];

  // preload slice 0 -> set0
#pragma unroll
  for (int i = 0; i < MT; ++i) a0[i] = load_a16x32(arow[i], hl);
#pragma unroll
  for (int j = 0; j < NT; ++j) b0[j] = load_b32x16(wrow[j], hl);

  int k = 0;
  for (; k + 64 < K; k += 64) {
    // slice k+32 -> set1
#pragma unroll
    for (int i = 0; i < MT; ++i) a1[i] = load_a16x32(arow[i] + k + 32, hl);
#pragma unroll
    for (int j = 0; j < NT; ++j) b1[j] = load_b32x16(wrow[j] + k + 32, hl);
    // consume set0 (slice k)
#pragma unroll
    for (int j = 0; j < NT; ++j)
#pragma unroll
      for (int i = 0; i < MT; ++i)
        acc[i][j] = wmma_bf16f32(a0[i], b0[j], acc[i][j]);
    // slice k+64 -> set0
#pragma unroll
    for (int i = 0; i < MT; ++i) a0[i] = load_a16x32(arow[i] + k + 64, hl);
#pragma unroll
    for (int j = 0; j < NT; ++j) b0[j] = load_b32x16(wrow[j] + k + 64, hl);
    // consume set1 (slice k+32)
#pragma unroll
    for (int j = 0; j < NT; ++j)
#pragma unroll
      for (int i = 0; i < MT; ++i)
        acc[i][j] = wmma_bf16f32(a1[i], b1[j], acc[i][j]);
  }
  // peel last pair: k == K-64 here
#pragma unroll
  for (int i = 0; i < MT; ++i) a1[i] = load_a16x32(arow[i] + k + 32, hl);
#pragma unroll
  for (int j = 0; j < NT; ++j) b1[j] = load_b32x16(wrow[j] + k + 32, hl);
#pragma unroll
  for (int j = 0; j < NT; ++j)
#pragma unroll
    for (int i = 0; i < MT; ++i)
      acc[i][j] = wmma_bf16f32(a0[i], b0[j], acc[i][j]);
#pragma unroll
  for (int j = 0; j < NT; ++j)
#pragma unroll
    for (int i = 0; i < MT; ++i)
      acc[i][j] = wmma_bf16f32(a1[i], b1[j], acc[i][j]);

#pragma unroll
  for (int j = 0; j < NT; ++j) {
    float badd = 0.f;
    if (bias1) badd += bias1[n0 + j * 16 + row];
    if (bias2) badd += bias2[n0 + j * 16 + row];
    const int n = n0 + j * 16 + row;
#pragma unroll
    for (int i = 0; i < MT; ++i) {
#pragma unroll
      for (int r = 0; r < 8; ++r) {
        const int m = m0 + i * 16 + (hl << 3) + r;
        out[(size_t)m * N + n] = acc[i][j][r] + badd;
      }
    }
  }
}

// ---------------------------------------------------------------------------
// One LSTM time step, fully fused: gates = xp[t] + h_{t-1} @ W_hh^T, then cell
// update. Each wave owns a 16-wide hidden-column group and computes the four
// gate tiles (i,f,g,o) so the cell nonlinearity is lane-local in the epilogue.
// Serial critical path (128 dependent launches, ~1 wave/SIMD): same manual
// ping-pong pipeline, no rotation copies.
// grid = 16 blocks x 128 threads = 64 waves = NH/16 column groups.
// ---------------------------------------------------------------------------
__global__ __launch_bounds__(128) void lstm_step(
    const bf16_t* __restrict__ hprev,   // (B=16, NH) bf16
    const bf16_t* __restrict__ Whh,     // (4NH, NH) bf16
    const float*  __restrict__ xp_t,    // (B, 4NH) f32 precomputed input proj
    float*        __restrict__ cstate,  // (B, NH) f32, updated in place
    bf16_t*       __restrict__ hnext,   // (B, NH) bf16 ping-pong
    float*        __restrict__ hseq_t)  // (B, NH) f32 output h_t
{
  const int lane = threadIdx.x & 31;
  const int wave = threadIdx.x >> 5;
  const int row  = lane & 15;
  const int hl   = lane >> 4;
  const int j    = ((blockIdx.x << 2) + wave) << 4;   // hidden column group

  const bf16_t* arow = hprev + (size_t)row * D_NH;
  const bf16_t* wr[4];
#pragma unroll
  for (int g = 0; g < 4; ++g)
    wr[g] = Whh + (size_t)(g * D_NH + j + row) * D_NH;

  v8f acc[4] = {};
  v16bf a0, b0[4], a1, b1[4];

  a0 = load_a16x32(arow, hl);
#pragma unroll
  for (int g = 0; g < 4; ++g) b0[g] = load_b32x16(wr[g], hl);

  int k = 0;
  for (; k + 64 < D_NH; k += 64) {
    a1 = load_a16x32(arow + k + 32, hl);
#pragma unroll
    for (int g = 0; g < 4; ++g) b1[g] = load_b32x16(wr[g] + k + 32, hl);
#pragma unroll
    for (int g = 0; g < 4; ++g) acc[g] = wmma_bf16f32(a0, b0[g], acc[g]);

    a0 = load_a16x32(arow + k + 64, hl);
#pragma unroll
    for (int g = 0; g < 4; ++g) b0[g] = load_b32x16(wr[g] + k + 64, hl);
#pragma unroll
    for (int g = 0; g < 4; ++g) acc[g] = wmma_bf16f32(a1, b1[g], acc[g]);
  }
  a1 = load_a16x32(arow + k + 32, hl);
#pragma unroll
  for (int g = 0; g < 4; ++g) b1[g] = load_b32x16(wr[g] + k + 32, hl);
#pragma unroll
  for (int g = 0; g < 4; ++g) acc[g] = wmma_bf16f32(a0, b0[g], acc[g]);
#pragma unroll
  for (int g = 0; g < 4; ++g) acc[g] = wmma_bf16f32(a1, b1[g], acc[g]);

  const int n = j + row;                 // hidden unit index
#pragma unroll
  for (int r = 0; r < 8; ++r) {
    const int m = (hl << 3) + r;         // batch index
    const float* xr = xp_t + (size_t)m * D_G4H;
    const float vi = acc[0][r] + xr[0 * D_NH + n];
    const float vf = acc[1][r] + xr[1 * D_NH + n];
    const float vg = acc[2][r] + xr[2 * D_NH + n];
    const float vo = acc[3][r] + xr[3 * D_NH + n];
    const size_t idx = (size_t)m * D_NH + n;
    const float c = sigf(vf) * cstate[idx] + sigf(vi) * tanhf(vg);
    cstate[idx] = c;
    const float h = sigf(vo) * tanhf(c);
    hseq_t[idx] = h;
    hnext[idx]  = (bf16_t)h;
  }
}

// ---------------------------------------------------------------------------
// Elementwise / setup kernels
// ---------------------------------------------------------------------------
__global__ void f32_to_bf16(bf16_t* __restrict__ dst, const float* __restrict__ src, int n) {
  int i = blockIdx.x * blockDim.x + threadIdx.x;
  if (i < n) dst[i] = (bf16_t)src[i];
}

// zf (B=16, NZ=256) bf16: [z (192) || label @ label_W^T (64)]
__global__ void build_zf(bf16_t* __restrict__ zf, const float* __restrict__ z,
                         const float* __restrict__ label, const float* __restrict__ labelW) {
  int i = blockIdx.x * 256 + threadIdx.x;
  if (i >= D_B * D_NZ) return;
  int b = i >> 8, k = i & 255;
  float v;
  if (k < D_NZ - D_LNZ) {
    v = z[b * (D_NZ - D_LNZ) + k];
  } else {
    int kk = k - (D_NZ - D_LNZ);
    v = label[b * 2 + 0] * labelW[kk * 2 + 0] + label[b * 2 + 1] * labelW[kk * 2 + 1];
  }
  zf[i] = (bf16_t)v;
}

// x (T*B, NI+NZ) bf16: [embedding gather || zf broadcast]
__global__ void build_x(bf16_t* __restrict__ x, const int* __restrict__ inputs,
                        const float* __restrict__ embedW, const bf16_t* __restrict__ zf) {
  long long i = (long long)blockIdx.x * 256 + threadIdx.x;
  if (i >= (long long)D_TB * D_XK) return;
  int tb = (int)(i / D_XK), k = (int)(i % D_XK);
  int b  = tb & (D_B - 1);
  if (k < D_NI) {
    int tok = inputs[tb];                       // inputs is (T,B) row-major
    x[i] = (bf16_t)embedW[(size_t)tok * D_NI + k];
  } else {
    x[i] = zf[b * D_NZ + (k - D_NI)];
  }
}

__global__ void h_init(bf16_t* __restrict__ h0, const float* __restrict__ c0, int n) {
  int i = blockIdx.x * blockDim.x + threadIdx.x;
  if (i < n) h0[i] = (bf16_t)tanhf(c0[i]);
}

// ---------------------------------------------------------------------------
// Attention + softmax(source dim) + context, writes bf16 cat row [h_t || ctx].
// One 128-thread block per (t,b) pair; LDS tree reductions for softmax.
// ---------------------------------------------------------------------------
__global__ __launch_bounds__(128) void attention_cat(
    bf16_t* __restrict__ cat,            // (T*B, 2NH) bf16
    const float* __restrict__ hseq,      // (T, B, NH) f32
    const float* __restrict__ hs)        // (S, B, NH) f32
{
  __shared__ float sc[D_SRC];
  __shared__ float red[128];
  const int tb  = blockIdx.x;
  const int b   = tb & (D_B - 1);
  const int tid = threadIdx.x;           // = source index s for phase 1

  const float* hvec = hseq + (size_t)tb * D_NH;
  const float* svec = hs + ((size_t)tid * D_B + b) * D_NH;
  float dot = 0.f;
  for (int h = 0; h < D_NH; ++h) dot += svec[h] * hvec[h];
  sc[tid] = dot;
  __syncthreads();

  // max reduce
  red[tid] = sc[tid];
  __syncthreads();
  for (int off = 64; off > 0; off >>= 1) {
    if (tid < off) red[tid] = fmaxf(red[tid], red[tid + off]);
    __syncthreads();
  }
  const float mx = red[0];
  __syncthreads();

  const float e = __expf(sc[tid] - mx);
  red[tid] = e;
  __syncthreads();
  for (int off = 64; off > 0; off >>= 1) {
    if (tid < off) red[tid] += red[tid + off];
    __syncthreads();
  }
  const float inv = 1.0f / red[0];
  __syncthreads();
  sc[tid] = e;                          // unnormalized attention weights
  __syncthreads();

  // context + cat row emit (bf16 for the final WMMA GEMM)
  bf16_t* crow = cat + (size_t)tb * D_2H;
#pragma unroll
  for (int q = 0; q < 8; ++q) {
    const int h = tid + 128 * q;
    float ctx = 0.f;
    for (int s = 0; s < D_SRC; ++s)
      ctx += sc[s] * hs[((size_t)s * D_B + b) * D_NH + h];
    crow[h]        = (bf16_t)hvec[h];
    crow[D_NH + h] = (bf16_t)(ctx * inv);
  }
}

// ---------------------------------------------------------------------------
// Host-side orchestration
// ---------------------------------------------------------------------------
extern "C" void kernel_launch(void* const* d_in, const int* in_sizes, int n_in,
                              void* d_out, int out_size, void* d_ws, size_t ws_size,
                              hipStream_t stream) {
  (void)in_sizes; (void)n_in; (void)out_size; (void)ws_size;
  const int*   inputs = (const int*)  d_in[0];
  const float* z      = (const float*)d_in[1];
  const float* label  = (const float*)d_in[2];
  const float* hs     = (const float*)d_in[3];
  const float* embedW = (const float*)d_in[4];
  const float* labelW = (const float*)d_in[5];
  const float* transW = (const float*)d_in[6];
  const float* W_ih   = (const float*)d_in[7];
  const float* W_hh   = (const float*)d_in[8];
  const float* b_ih   = (const float*)d_in[9];
  const float* b_hh   = (const float*)d_in[10];
  const float* predW  = (const float*)d_in[11];

  char* ws = (char*)d_ws;
  size_t off = 0;
  auto carve = [&](size_t bytes) -> char* {
    char* p = ws + off;
    off = (off + bytes + 255) & ~(size_t)255;
    return p;
  };

  bf16_t* zf      = (bf16_t*)carve((size_t)D_B * D_NZ * 2);
  bf16_t* xb      = (bf16_t*)carve((size_t)D_TB * D_XK * 2);
  bf16_t* Wih_b   = (bf16_t*)carve((size_t)D_G4H * D_XK * 2);
  bf16_t* Whh_b   = (bf16_t*)carve((size_t)D_G4H * D_NH * 2);
  bf16_t* Wtr_b   = (bf16_t*)carve((size_t)D_NH * D_NZ * 2);
  bf16_t* Wpred_b = (bf16_t*)carve((size_t)D_V * D_2H * 2);
  float*  xp      = (float*) carve((size_t)D_TB * D_G4H * 4);
  float*  cstate  = (float*) carve((size_t)D_B * D_NH * 4);
  bf16_t* hping0  = (bf16_t*)carve((size_t)D_B * D_NH * 2);
  bf16_t* hping1  = (bf16_t*)carve((size_t)D_B * D_NH * 2);
  float*  hseq    = (float*) carve((size_t)D_TB * D_NH * 4);
  bf16_t* cat     = (bf16_t*)carve((size_t)D_TB * D_2H * 2);
  bf16_t* hping[2] = { hping0, hping1 };

  auto cvt = [&](bf16_t* dst, const float* src, int n) {
    f32_to_bf16<<<(n + 255) / 256, 256, 0, stream>>>(dst, src, n);
  };

  // 1) weight precision conversion (one-time per launch, amortized over GEMMs)
  cvt(Wih_b,   W_ih,   D_G4H * D_XK);
  cvt(Whh_b,   W_hh,   D_G4H * D_NH);
  cvt(Wtr_b,   transW, D_NH * D_NZ);
  cvt(Wpred_b, predW,  D_V * D_2H);

  // 2) latent vector and LSTM input assembly
  build_zf<<<(D_B * D_NZ + 255) / 256, 256, 0, stream>>>(zf, z, label, labelW);
  build_x<<<(int)(((long long)D_TB * D_XK + 255) / 256), 256, 0, stream>>>(
      xb, inputs, embedW, zf);

  // 3) xp = x @ W_ih^T + b_ih + b_hh  (M=2048, N=4096, K=768), 64x256 blocks
  {
    dim3 g(D_G4H / 256, D_TB / 64);
    gemm_bf16_wmma<4, 4><<<g, 128, 0, stream>>>(xp, xb, Wih_b, b_ih, b_hh,
                                                D_TB, D_G4H, D_XK);
  }

  // 4) c0 = zf @ trans_W^T (M=16, N=1024, K=256); h0 = tanh(c0)
  {
    dim3 g(D_NH / 256, 1);
    gemm_bf16_wmma<1, 4><<<g, 128, 0, stream>>>(cstate, zf, Wtr_b, nullptr,
                                                nullptr, D_B, D_NH, D_NZ);
  }
  h_init<<<(D_B * D_NH + 255) / 256, 256, 0, stream>>>(hping[0], cstate, D_B * D_NH);

  // 5) sequential LSTM recurrence, one fused kernel per step
  for (int t = 0; t < D_SEQ; ++t) {
    lstm_step<<<16, 128, 0, stream>>>(hping[t & 1], Whh_b,
                                      xp + (size_t)t * D_B * D_G4H, cstate,
                                      hping[(t + 1) & 1],
                                      hseq + (size_t)t * D_B * D_NH);
  }

  // 6) attention + softmax + context -> cat rows (bf16)
  attention_cat<<<D_TB, 128, 0, stream>>>(cat, hseq, hs);

  // 7) logits = cat @ pred_W^T (M=2048, N=32000, K=2048) -> d_out (f32)
  //    64M x 64N wave tiles balance A/B L2 traffic at ~4 GB each.
  {
    dim3 g(D_V / 256, D_TB / 64);
    gemm_bf16_wmma<4, 4><<<g, 128, 0, stream>>>((float*)d_out, cat, Wpred_b,
                                                nullptr, nullptr, D_TB, D_V, D_2H);
  }
}
